// DeformConvModule_1219770712155
// MI455X (gfx1250) — compile-verified, matrix operand order checked by reading, not run on hardware
//
#include <hip/hip_runtime.h>
#include <cstddef>

typedef __bf16 v16bf __attribute__((ext_vector_type(16)));
typedef float  v8f   __attribute__((ext_vector_type(8)));

#define BATCH 4
#define CIN   256
#define HH    64
#define WW    64
#define OUTC  256
#define HWSZ  (HH*WW)          // 4096
#define NTOT  (BATCH*HWSZ)     // 16384
#define KDIM  (9*CIN)          // 2304
#define NKT   (KDIM/32)        // 72 K-chunks of 32
#define NNT   (NTOT/16)        // 1024 N tiles
#define NMT   (OUTC/16)        // 16 M tiles

// ---------------------------------------------------------------------------
// Kernel 1: offset conv (18 channels, 3x3, pad 1). One thread per pixel,
// 18 fp32 accumulators; weight loads are wave-uniform -> scalar path.
// off layout: [b][j][hw], j = k2*2 + {0:y,1:x}
// ---------------------------------------------------------------------------
__global__ void __launch_bounds__(256)
off_conv_kernel(const float* __restrict__ x, const float* __restrict__ w_off,
                const float* __restrict__ b_off, float* __restrict__ off) {
    int n  = blockIdx.x * 256 + threadIdx.x;       // 0..16383
    int b  = n >> 12;
    int hw = n & (HWSZ - 1);
    int h  = hw >> 6, w = hw & 63;

    float acc[18];
#pragma unroll
    for (int j = 0; j < 18; ++j) acc[j] = b_off[j];

    const float* xb = x + (size_t)b * CIN * HWSZ;
    for (int c = 0; c < CIN; ++c) {
        const float* xc = xb + c * HWSZ;
#pragma unroll
        for (int dy = 0; dy < 3; ++dy) {
            int yy = h + dy - 1;
            if (yy < 0 || yy > HH - 1) continue;
#pragma unroll
            for (int dx = 0; dx < 3; ++dx) {
                int xx = w + dx - 1;
                if (xx < 0 || xx > WW - 1) continue;
                float xv = xc[yy * WW + xx];
                const float* wp = w_off + (c * 9 + dy * 3 + dx); // [j][c][3][3]
#pragma unroll
                for (int j = 0; j < 18; ++j)
                    acc[j] = fmaf(wp[j * (CIN * 9)], xv, acc[j]);
            }
        }
    }
#pragma unroll
    for (int j = 0; j < 18; ++j)
        off[((size_t)b * 18 + j) * HWSZ + hw] = acc[j];
}

// ---------------------------------------------------------------------------
// Kernel 2: pack w_def -> bf16 A-matrix tiles in the per-lane WMMA layout.
// Apack[mt][kt][lane][16], K order: kd = tap*256 + c.
// ---------------------------------------------------------------------------
__global__ void __launch_bounds__(256)
pack_a_kernel(const float* __restrict__ w_def, __bf16* __restrict__ Apack) {
    int idx = blockIdx.x * 256 + threadIdx.x;      // NMT*NKT*32 = 36864
    if (idx >= NMT * NKT * 32) return;
    int lane = idx & 31;
    int kt   = (idx >> 5) % NKT;
    int mt   = idx / (NKT * 32);
    int half = lane >> 4;
    int m    = mt * 16 + (lane & 15);              // output channel o
    int kk   = kt >> 3;                            // tap 0..8
    int cg   = kt & 7;                             // channel group of 32

    union { __bf16 e[16]; v16bf v; } u;
#pragma unroll
    for (int j = 0; j < 16; ++j) {
        int kl = (j < 8) ? (half * 8 + j) : (16 + half * 8 + (j - 8));
        int c  = cg * 32 + kl;
        u.e[j] = (__bf16)w_def[((size_t)m * CIN + c) * 9 + kk];
    }
    *reinterpret_cast<v16bf*>(Apack + (size_t)idx * 16) = u.v;
}

// ---------------------------------------------------------------------------
// Kernel 3: bilinear gather + bf16 B-pack.
// One block = 16 pixels x 1 tap x all 256 channels. Corner offsets/weights
// computed once per pixel into LDS; x planes are L2-resident (16.7 MB).
// Bpack[nt][kt][lane][16] with same per-lane layout / K order as Apack.
// ---------------------------------------------------------------------------
__global__ void __launch_bounds__(256)
gather_pack_kernel(const float* __restrict__ x, const float* __restrict__ off,
                   __bf16* __restrict__ Bpack) {
    int bid = blockIdx.x;            // NNT*9 = 9216 blocks
    int kk  = bid % 9;
    int nt  = bid / 9;
    int ky  = kk / 3, kx = kk % 3;
    int t   = threadIdx.x;

    __shared__ int    s_off[16][4];
    __shared__ float  s_w[16][4];
    __shared__ int    s_b[16];
    __shared__ __bf16 stage[CIN * 16];   // [c][n_local], 8 KB

    if (t < 16) {
        int n  = nt * 16 + t;
        int b  = n >> 12;
        int hw = n & (HWSZ - 1);
        int h  = hw >> 6, w = hw & 63;
        float py = off[((size_t)b * 18 + kk * 2 + 0) * HWSZ + hw] + (float)(ky + h - 1);
        float px = off[((size_t)b * 18 + kk * 2 + 1) * HWSZ + hw] + (float)(kx + w - 1);
        float y0f = floorf(py), x0f = floorf(px);
        float wy1 = py - y0f, wy0 = 1.0f - wy1;
        float wx1 = px - x0f, wx0 = 1.0f - wx1;
        int   y0  = (int)y0f, x0 = (int)x0f;
        int   ys[2] = { y0, y0 + 1 };
        int   xs[2] = { x0, x0 + 1 };
        float wys[2] = { wy0, wy1 };
        float wxs[2] = { wx0, wx1 };
#pragma unroll
        for (int i = 0; i < 4; ++i) {
            int yy = ys[i >> 1], xx = xs[i & 1];
            bool valid = (yy >= 0) && (yy <= HH - 1) && (xx >= 0) && (xx <= WW - 1);
            int yc = min(max(yy, 0), HH - 1);
            int xc = min(max(xx, 0), WW - 1);
            s_off[t][i] = yc * WW + xc;
            s_w[t][i]   = valid ? (wys[i >> 1] * wxs[i & 1]) : 0.0f;
        }
        s_b[t] = b;
    }
    __syncthreads();

    // Phase 2: thread t = channel c; 16 pixels each. LDS reads are broadcasts.
    {
        int c = t;
#pragma unroll 4
        for (int nl = 0; nl < 16; ++nl) {
            const float* xb = x + ((size_t)(s_b[nl] * CIN + c)) * HWSZ;
            float v = s_w[nl][0] * xb[s_off[nl][0]]
                    + s_w[nl][1] * xb[s_off[nl][1]]
                    + s_w[nl][2] * xb[s_off[nl][2]]
                    + s_w[nl][3] * xb[s_off[nl][3]];
            stage[c * 16 + nl] = (__bf16)v;
        }
    }
    __syncthreads();

    // Phase 3: coalesced packed writes; thread = (cg, lane), 32 B each.
    {
        int cg   = t >> 5;           // 0..7
        int lane = t & 31;
        int half = lane >> 4;
        int nl   = lane & 15;
        union { __bf16 e[16]; v16bf v; } u;
#pragma unroll
        for (int j = 0; j < 16; ++j) {
            int kl = (j < 8) ? (half * 8 + j) : (16 + half * 8 + (j - 8));
            int c  = cg * 32 + kl;
            u.e[j] = stage[c * 16 + nl];
        }
        size_t base = (((size_t)nt * NKT + (kk * 8 + cg)) * 32 + lane) * 16;
        *reinterpret_cast<v16bf*>(Bpack + base) = u.v;
    }
}

// ---------------------------------------------------------------------------
// Kernel 4: WMMA GEMM with LDS-staged, double-buffered B tiles.
// Block = 8 waves = 8 M-tiles x 8 N-tiles (shared B). Steady-state chunk:
//   - 256 threads prefetch next 8KB B chunk unconditionally (1 v16bf each)
//   - each wave: 1 global A prefetch + 8 LDS B loads + 8 v_wmma
//   - one barrier per chunk (double-buffered LDS); last chunk peeled
// Grid: 2 (M halves) x 128 (N groups) = 256 blocks, EXEC always full.
// ---------------------------------------------------------------------------
__device__ __forceinline__ void store_dtile(v8f acc, float* __restrict__ out,
                                            int mt, int half, int nl, int n_base) {
    int n  = n_base + nl;
    int b  = n >> 12;
    int hw = n & (HWSZ - 1);
    float* op = out + ((size_t)(b * OUTC + mt * 16 + half * 8)) * HWSZ + hw;
#pragma unroll
    for (int r = 0; r < 8; ++r) op[(size_t)r * HWSZ] = acc[r];
}

__global__ void __launch_bounds__(256)
wmma_gemm_kernel(const __bf16* __restrict__ Apack, const __bf16* __restrict__ Bpack,
                 float* __restrict__ out) {
    __shared__ __bf16 ldsB[2][8][32][16];   // 2 x 8KB double buffer

    int t    = threadIdx.x;
    int lane = t & 31;
    int wv   = t >> 5;                      // wave id = local M tile (0..7)
    int mh   = blockIdx.x & 1;              // M half
    int ng   = blockIdx.x >> 1;             // N group (0..127)
    int mt   = mh * 8 + wv;
    int nt0  = ng * 8;

    const v16bf* Bg = reinterpret_cast<const v16bf*>(Bpack)
                    + ((size_t)(nt0 + wv) * NKT) * 32 + lane;   // coop-load slot
    const v16bf* Ag = reinterpret_cast<const v16bf*>(Apack)
                    + (size_t)mt * NKT * 32 + lane;

    // Preload chunk 0 into buffer 0.
    *reinterpret_cast<v16bf*>(&ldsB[0][wv][lane][0]) = Bg[0];
    __syncthreads();

    v8f acc0 = {}, acc1 = {}, acc2 = {}, acc3 = {};
    v8f acc4 = {}, acc5 = {}, acc6 = {}, acc7 = {};
    v16bf a = Ag[0];

#define WMMA8(BASE)                                                            \
    do {                                                                       \
        const __bf16* bb = (BASE);                                             \
        v16bf b0 = *reinterpret_cast<const v16bf*>(bb + 0 * 512);              \
        v16bf b1 = *reinterpret_cast<const v16bf*>(bb + 1 * 512);              \
        v16bf b2 = *reinterpret_cast<const v16bf*>(bb + 2 * 512);              \
        v16bf b3 = *reinterpret_cast<const v16bf*>(bb + 3 * 512);              \
        v16bf b4 = *reinterpret_cast<const v16bf*>(bb + 4 * 512);              \
        v16bf b5 = *reinterpret_cast<const v16bf*>(bb + 5 * 512);              \
        v16bf b6 = *reinterpret_cast<const v16bf*>(bb + 6 * 512);              \
        v16bf b7 = *reinterpret_cast<const v16bf*>(bb + 7 * 512);              \
        acc0 = __builtin_amdgcn_wmma_f32_16x16x32_bf16(false, a, false, b0, (short)0, acc0, false, false); \
        acc1 = __builtin_amdgcn_wmma_f32_16x16x32_bf16(false, a, false, b1, (short)0, acc1, false, false); \
        acc2 = __builtin_amdgcn_wmma_f32_16x16x32_bf16(false, a, false, b2, (short)0, acc2, false, false); \
        acc3 = __builtin_amdgcn_wmma_f32_16x16x32_bf16(false, a, false, b3, (short)0, acc3, false, false); \
        acc4 = __builtin_amdgcn_wmma_f32_16x16x32_bf16(false, a, false, b4, (short)0, acc4, false, false); \
        acc5 = __builtin_amdgcn_wmma_f32_16x16x32_bf16(false, a, false, b5, (short)0, acc5, false, false); \
        acc6 = __builtin_amdgcn_wmma_f32_16x16x32_bf16(false, a, false, b6, (short)0, acc6, false, false); \
        acc7 = __builtin_amdgcn_wmma_f32_16x16x32_bf16(false, a, false, b7, (short)0, acc7, false, false); \
    } while (0)

    // Steady state: unconditional prefetch of chunk kt+1, consume chunk kt.
#pragma unroll 2
    for (int kt = 0; kt < NKT - 1; ++kt) {
        v16bf breg = Bg[(size_t)(kt + 1) * 32];
        v16bf a_n  = Ag[(size_t)(kt + 1) * 32];
        WMMA8(&ldsB[kt & 1][0][lane][0]);
        *reinterpret_cast<v16bf*>(&ldsB[(kt + 1) & 1][wv][lane][0]) = breg;
        a = a_n;
        __syncthreads();
    }
    // Peeled last chunk: no prefetch, no barrier.
    WMMA8(&ldsB[(NKT - 1) & 1][0][lane][0]);
#undef WMMA8

    int half = lane >> 4;
    int nl   = lane & 15;
    store_dtile(acc0, out, mt, half, nl, (nt0 + 0) * 16);
    store_dtile(acc1, out, mt, half, nl, (nt0 + 1) * 16);
    store_dtile(acc2, out, mt, half, nl, (nt0 + 2) * 16);
    store_dtile(acc3, out, mt, half, nl, (nt0 + 3) * 16);
    store_dtile(acc4, out, mt, half, nl, (nt0 + 4) * 16);
    store_dtile(acc5, out, mt, half, nl, (nt0 + 5) * 16);
    store_dtile(acc6, out, mt, half, nl, (nt0 + 6) * 16);
    store_dtile(acc7, out, mt, half, nl, (nt0 + 7) * 16);
}

// ---------------------------------------------------------------------------
// Kernel 5: BN statistics per output channel (biased var, matches jnp.var).
// ---------------------------------------------------------------------------
__global__ void __launch_bounds__(256)
bn_stats_kernel(const float* __restrict__ out, float* __restrict__ mv) {
    int o = blockIdx.x;        // 256
    int t = threadIdx.x;       // 256
    float s = 0.0f, ss = 0.0f;
    for (int i = t; i < NTOT; i += 256) {
        int b  = i >> 12;
        int hw = i & (HWSZ - 1);
        float v = out[((size_t)(b * OUTC + o)) * HWSZ + hw];
        s += v; ss += v * v;
    }
    __shared__ float sh_s[256], sh_q[256];
    sh_s[t] = s; sh_q[t] = ss;
    __syncthreads();
    for (int step = 128; step > 0; step >>= 1) {
        if (t < step) { sh_s[t] += sh_s[t + step]; sh_q[t] += sh_q[t + step]; }
        __syncthreads();
    }
    if (t == 0) {
        float mean = sh_s[0] * (1.0f / NTOT);
        float var  = sh_q[0] * (1.0f / NTOT) - mean * mean;
        mv[o]        = mean;
        mv[OUTC + o] = var;
    }
}

// ---------------------------------------------------------------------------
// Kernel 6: BN apply + ReLU, in-place on d_out.
// ---------------------------------------------------------------------------
__global__ void __launch_bounds__(256)
bn_apply_kernel(float* __restrict__ out, const float* __restrict__ mv,
                const float* __restrict__ gamma, const float* __restrict__ beta) {
    int idx = blockIdx.x * 256 + threadIdx.x;       // 4*256*4096
    if (idx >= BATCH * OUTC * HWSZ) return;
    int o = (idx >> 12) & (OUTC - 1);
    float scale = gamma[o] * rsqrtf(mv[OUTC + o] + 1e-5f);
    float v = (out[idx] - mv[o]) * scale + beta[o];
    out[idx] = v > 0.0f ? v : 0.0f;
}

// ---------------------------------------------------------------------------
extern "C" void kernel_launch(void* const* d_in, const int* in_sizes, int n_in,
                              void* d_out, int out_size, void* d_ws, size_t ws_size,
                              hipStream_t stream) {
    const float* x     = (const float*)d_in[0];
    const float* w_off = (const float*)d_in[1];
    const float* b_off = (const float*)d_in[2];
    const float* w_def = (const float*)d_in[3];
    const float* gamma = (const float*)d_in[4];
    const float* beta  = (const float*)d_in[5];
    float* out = (float*)d_out;

    // Workspace carve (all 256B-aligned):
    //   off   : 4*18*4096 f32        = 1,179,648 B
    //   Apack : 16*72*512 bf16       = 1,179,648 B
    //   Bpack : 16384*2304 bf16      = 75,497,472 B
    //   mv    : 512 f32              = 2,048 B      (total ~77.9 MB)
    char*   ws    = (char*)d_ws;
    float*  offb  = (float*)ws;
    __bf16* Apack = (__bf16*)(ws + 1179648);
    __bf16* Bpack = (__bf16*)(ws + 1179648 + 1179648);
    float*  mv    = (float*)(ws + 1179648 + 1179648 + 75497472);

    off_conv_kernel<<<NTOT / 256, 256, 0, stream>>>(x, w_off, b_off, offb);
    pack_a_kernel<<<(NMT * NKT * 32 + 255) / 256, 256, 0, stream>>>(w_def, Apack);
    gather_pack_kernel<<<NNT * 9, 256, 0, stream>>>(x, offb, Bpack);
    wmma_gemm_kernel<<<2 * (NNT / 8), 256, 0, stream>>>(Apack, Bpack, out);
    bn_stats_kernel<<<OUTC, 256, 0, stream>>>(out, mv);
    bn_apply_kernel<<<(BATCH * OUTC * HWSZ) / 256, 256, 0, stream>>>(out, mv, gamma, beta);
}
